// SymmetricContraction_81879256531533
// MI455X (gfx1250) — matrix operand order repaired and checked by reading, not compiled, not run on hardware
//
#include <hip/hip_runtime.h>

typedef __attribute__((ext_vector_type(16))) _Float16 v16h;
typedef __attribute__((ext_vector_type(8)))  float    v8f;

#define B_N 1024
#define C_N 128
#define E_N 10
#define L_D 16
#define K3 23
#define K2 4
#define NCHUNK 12   // ceil(368/32) -> K padded to 384
#define NRT 8       // 128 channels / 16 rows per tile
#define TPB 256     // 8 wave32

__global__ __launch_bounds__(TPB, 1)
void symcontract_kernel(const float* __restrict__ xg,
                        const float* __restrict__ yg,
                        const float* __restrict__ U3g,
                        const float* __restrict__ U2g,
                        const float* __restrict__ U1g,
                        const float* __restrict__ W3g,
                        const float* __restrict__ W2g,
                        const float* __restrict__ W1g,
                        float* __restrict__ outg)
{
    // LDS: A fragments (wmma layout, f16) + per-node staging. ~126 KB total.
    __shared__ __align__(32) _Float16 A_sh[NRT * NCHUNK * 32 * 16]; // 96 KB
    __shared__ float x_sh[C_N * L_D];     // 8 KB
    __shared__ float w3_sh[K3 * C_N];     // 11.5 KB
    __shared__ float w2_sh[K2 * C_N];     // 2 KB
    __shared__ float w1_sh[C_N];          // 0.5 KB
    __shared__ float s_sh[L_D * C_N];     // 8 KB

    const int tid  = threadIdx.x;
    const int lane = tid & 31;
    const int wave = tid >> 5;     // 0..7
    const int xcol = lane & 15;    // N index (x) within 16x16 tile
    const int hi   = lane >> 4;    // lane half

    // ---- Register-resident B: U3 reshaped [K=368 -> pad 384, wx=256], f16.
    // B 32x16 in contiguous k order (same as A): lanes 0-15 hold K=0..15 (elem e -> K), lanes 16-31 K=16..31.
    v16h breg[2][NCHUNK];
    #pragma unroll
    for (int j = 0; j < 2; ++j) {
        const int w = wave + j * 8;
        #pragma unroll
        for (int ck = 0; ck < NCHUNK; ++ck) {
            const int k = ck * 2 + hi;        // K>>4 for this lane-half/chunk
            v16h bv;
            #pragma unroll
            for (int e = 0; e < 16; ++e) {    // i = e (K & 15)
                float v = (k < K3) ? U3g[((w * L_D + xcol) * L_D + e) * K3 + k] : 0.f;
                bv[e] = (_Float16)v;
            }
            breg[j][ck] = bv;
        }
    }
    // Per-lane U2 slice for the nu=2 epilogue.
    float u2reg[2][K2];
    #pragma unroll
    for (int j = 0; j < 2; ++j) {
        const int w = wave + j * 8;
        #pragma unroll
        for (int k = 0; k < K2; ++k)
            u2reg[j][k] = U2g[(w * L_D + xcol) * K2 + k];
    }

    for (int b = blockIdx.x; b < B_N; b += gridDim.x) {
        // ---- Phase 1: stage x, mix element weights w3/w2/w1 = W? . y[b]
        float yreg[E_N];
        #pragma unroll
        for (int e = 0; e < E_N; ++e) yreg[e] = yg[b * E_N + e];

        for (int idx = tid; idx < C_N * L_D / 4; idx += TPB)
            ((float4*)x_sh)[idx] = ((const float4*)xg)[b * (C_N * L_D / 4) + idx];

        for (int idx = tid; idx < K3 * C_N; idx += TPB) {
            float s = 0.f;
            #pragma unroll
            for (int e = 0; e < E_N; ++e) s += yreg[e] * W3g[e * K3 * C_N + idx];
            w3_sh[idx] = s;
        }
        for (int idx = tid; idx < K2 * C_N; idx += TPB) {
            float s = 0.f;
            #pragma unroll
            for (int e = 0; e < E_N; ++e) s += yreg[e] * W2g[e * K2 * C_N + idx];
            w2_sh[idx] = s;
        }
        for (int idx = tid; idx < C_N; idx += TPB) {
            float s = 0.f;
            #pragma unroll
            for (int e = 0; e < E_N; ++e) s += yreg[e] * W1g[e * C_N + idx];
            w1_sh[idx] = s;
        }
        __syncthreads();

        // ---- Phase 2: build A fragments (t[bc, k*16+i] = w3[k,c]*x[c,i])
        // in 16x32 A layout, contiguous k (K = 16*half + e); one 16-elem lane-chunk per trip.
        for (int trip = tid; trip < NRT * NCHUNK * 32; trip += TPB) {
            const int ln = trip & 31;
            const int ck = (trip >> 5) % NCHUNK;
            const int rt = trip / (NCHUNK * 32);
            const int c  = rt * 16 + (ln & 15);          // M = row = channel
            v16h av;
            #pragma unroll
            for (int e = 0; e < 16; ++e) {
                const int K = ck * 32 + ((ln >> 4) << 4)
                            + e;
                const int k = K >> 4, i = K & 15;
                av[e] = (k < K3) ? (_Float16)(w3_sh[k * C_N + c] * x_sh[c * L_D + i])
                                 : (_Float16)0.f;
            }
            *(v16h*)(&A_sh[trip * 16]) = av;
        }
        __syncthreads();

        // ---- Phase 3: GEMM via WMMA + fused nu=2 epilogue -> s[w][c]
        for (int rt = 0; rt < NRT; ++rt) {
            v8f acc0 = {}; v8f acc1 = {};
            #pragma unroll
            for (int ck = 0; ck < NCHUNK; ++ck) {
                v16h a = *(const v16h*)(&A_sh[((rt * NCHUNK + ck) * 32 + lane) * 16]);
                acc0 = __builtin_amdgcn_wmma_f32_16x16x32_f16(
                           false, a, false, breg[0][ck], (short)0, acc0, false, false);
                acc1 = __builtin_amdgcn_wmma_f32_16x16x32_f16(
                           false, a, false, breg[1][ck], (short)0, acc1, false, false);
            }
            #pragma unroll
            for (int j = 0; j < 2; ++j) {
                v8f acc = j ? acc1 : acc0;
                const int w = wave + j * 8;
                #pragma unroll
                for (int r = 0; r < 8; ++r) {
                    const int c = rt * 16 + r + (hi << 3);   // M = r + 8*half
                    float q = acc[r];
                    #pragma unroll
                    for (int k = 0; k < K2; ++k)
                        q += u2reg[j][k] * w2_sh[k * C_N + c]; // + U2.w2
                    float val = q * x_sh[c * L_D + xcol];      // * x_x
                    val += __shfl_xor(val, 1);                 // reduce over 16 x-lanes
                    val += __shfl_xor(val, 2);
                    val += __shfl_xor(val, 4);
                    val += __shfl_xor(val, 8);
                    if (xcol == 0) s_sh[w * C_N + c] = val;
                }
            }
        }
        __syncthreads();

        // ---- Phase 4: out[b,c] = sum_w (s[w][c] + U1[w]*w1[c]) * x[c][w]
        if (tid < C_N) {
            const int c = tid;
            const float w1c = w1_sh[c];
            float o = 0.f;
            #pragma unroll
            for (int w = 0; w < L_D; ++w) {
                const float c1 = s_sh[w * C_N + c] + U1g[w] * w1c;
                o += c1 * x_sh[c * L_D + w];
            }
            outg[b * C_N + c] = o;
        }
        __syncthreads();   // protect LDS before next node
    }
}

extern "C" void kernel_launch(void* const* d_in, const int* in_sizes, int n_in,
                              void* d_out, int out_size, void* d_ws, size_t ws_size,
                              hipStream_t stream) {
    const float* x  = (const float*)d_in[0];
    const float* y  = (const float*)d_in[1];
    const float* U3 = (const float*)d_in[2];
    const float* U2 = (const float*)d_in[3];
    const float* U1 = (const float*)d_in[4];
    const float* W3 = (const float*)d_in[5];
    const float* W2 = (const float*)d_in[6];
    const float* W1 = (const float*)d_in[7];
    float* out = (float*)d_out;
    symcontract_kernel<<<256, TPB, 0, stream>>>(x, y, U3, U2, U1, W3, W2, W1, out);
}
